// SelfAttention_78975858638937
// MI455X (gfx1250) — matrix operand order repaired and checked by reading, not compiled, not run on hardware
//
#include <hip/hip_runtime.h>

// ---------------------------------------------------------------------------
// Problem constants (match reference)
// ---------------------------------------------------------------------------
constexpr int D_MODEL = 1024;
constexpr int NH      = 16;
constexpr int HD      = 64;     // head dim
constexpr int BB      = 4;      // batch
constexpr int SS      = 2048;   // seq len
constexpr int MROWS   = BB * SS;        // 8192
constexpr float NEGF  = -1e10f;

typedef __attribute__((ext_vector_type(16))) __bf16 v16bf;
typedef __attribute__((ext_vector_type(8)))  __bf16 v8bf;
typedef __attribute__((ext_vector_type(8)))  float  v8f;

// ---------------------------------------------------------------------------
// bf16 helpers (manual RNE convert; __bf16 used as storage type only)
// ---------------------------------------------------------------------------
static __device__ __forceinline__ __bf16 f2bf(float f) {
    unsigned u = __builtin_bit_cast(unsigned, f);
    unsigned r = u + 0x7FFFu + ((u >> 16) & 1u);
    unsigned short h = (unsigned short)(r >> 16);
    return __builtin_bit_cast(__bf16, h);
}

static __device__ __forceinline__ void store_out(float* p, float v)  { *p = v; }
static __device__ __forceinline__ void store_out(__bf16* p, float v) { *p = f2bf(v); }

// Load a 16-bit A fragment (16x32, row-major source, contiguous k order):
// lane m = lane&15, kb16 = (lane&16)?16:0 ; elems j:0..15 -> K=kb16+j
static __device__ __forceinline__ v16bf load_a_frag(const __bf16* p) {
    v8bf lo = *(const v8bf*)p;
    v8bf hi = *(const v8bf*)(p + 8);
    v16bf a;
#pragma unroll
    for (int j = 0; j < 8; ++j) { a[j] = lo[j]; a[8 + j] = hi[j]; }
    return a;
}

// ---------------------------------------------------------------------------
// Elementwise converters
// ---------------------------------------------------------------------------
__global__ void cvt_f32_bf16(const float* __restrict__ in, __bf16* __restrict__ out, int n) {
    int i = blockIdx.x * blockDim.x + threadIdx.x;
    if (i < n) out[i] = f2bf(in[i]);
}

// Wt[n, k] = W[k, n]  (1024x1024)
__global__ void transpose_cvt(const float* __restrict__ W, __bf16* __restrict__ Wt) {
    int i = blockIdx.x * blockDim.x + threadIdx.x;   // i = n*1024 + k
    int n = i >> 10;
    int k = i & 1023;
    Wt[i] = f2bf(W[k * D_MODEL + n]);
}

// ---------------------------------------------------------------------------
// bf16 GEMM: C[M,N] = A[M,K] @ Bt[N,K]^T + bias[N]
//   block = 256 thr (8 waves) -> 256(M) x 64(N) tile.
//   wave = 64(M) x 32(N): 4 M-tiles x 2 N-tiles = 8 WMMA accumulators,
//   8 WMMAs per 12 b128 loads per K-step (compute-bound regime).
//   TRANS_V: scatter output into Vt[B][D_MODEL][SS] (V stored transposed per head)
// ---------------------------------------------------------------------------
template <typename OutT, bool TRANS_V>
__global__ __launch_bounds__(256)
void gemm_bf16(const __bf16* __restrict__ A, const __bf16* __restrict__ Bt,
               const float* __restrict__ bias, OutT* __restrict__ C,
               int M, int N, int K) {
    const int lane = threadIdx.x & 31;
    const int wave = threadIdx.x >> 5;
    const int wn = wave & 1;            // 0..1  (32 cols each)
    const int wm = wave >> 1;           // 0..3  (64 rows each)
    const int n0 = blockIdx.x * 64 + wn * 32;
    const int m0 = blockIdx.y * 256 + wm * 64;

    const int ml   = lane & 15;
    const int kb8  = (lane & 16) ? 8  : 0;
    const int kb16 = (lane & 16) ? 16 : 0;
    const int hi8  = (lane & 16) ? 8  : 0;

    v8f acc[8];   // [t*2 + u] : t = M-tile 0..3, u = N-tile 0..1
#pragma unroll
    for (int i = 0; i < 8; ++i)
#pragma unroll
        for (int r = 0; r < 8; ++r) acc[i][r] = 0.f;

    const __bf16* bp0 = Bt + (size_t)(n0 + ml) * K + kb16;
    const __bf16* bp1 = bp0 + (size_t)16 * K;

    for (int k0 = 0; k0 < K; k0 += 32) {
        v16bf b0 = *(const v16bf*)(bp0 + k0);
        v16bf b1 = *(const v16bf*)(bp1 + k0);
        if (k0 + 64 < K)  // stream the A panel ahead (global_prefetch_b8)
            __builtin_prefetch(A + (size_t)(m0 + ml) * K + k0 + 64, 0, 3);
#pragma unroll
        for (int t = 0; t < 4; ++t) {
            const __bf16* ap = A + (size_t)(m0 + t * 16 + ml) * K + k0 + kb16;
            v16bf af = load_a_frag(ap);
            acc[t * 2 + 0] = __builtin_amdgcn_wmma_f32_16x16x32_bf16(
                false, af, false, b0, (short)0, acc[t * 2 + 0], false, false);
            acc[t * 2 + 1] = __builtin_amdgcn_wmma_f32_16x16x32_bf16(
                false, af, false, b1, (short)0, acc[t * 2 + 1], false, false);
        }
    }

    const float bval0 = bias[n0 + ml];
    const float bval1 = bias[n0 + 16 + ml];
#pragma unroll
    for (int t = 0; t < 4; ++t) {
#pragma unroll
        for (int u = 0; u < 2; ++u) {
            const int col = n0 + u * 16 + ml;
            const float bval = u ? bval1 : bval0;
#pragma unroll
            for (int r = 0; r < 8; ++r) {
                int row = m0 + t * 16 + hi8 + r;
                float v = acc[t * 2 + u][r] + bval;
                if constexpr (TRANS_V) {
                    int b = row >> 11;          // row / SS
                    int s = row & (SS - 1);     // row % SS
                    store_out(&C[((size_t)b * D_MODEL + col) * SS + s], v);
                } else {
                    store_out(&C[(size_t)row * N + col], v);
                }
            }
        }
    }
}

// ---------------------------------------------------------------------------
// Flash attention: one wave per (b, h, 16-query tile). Streams 32 keys/iter.
//   Q,K: bf16 [B*S, D_MODEL] (head h at cols h*64..)
//   Vt : bf16 [B][D_MODEL][SS] (transposed per head)
//   Ctx: bf16 [B*S, D_MODEL]
// ---------------------------------------------------------------------------
__global__ __launch_bounds__(32)
void attn_kernel(const __bf16* __restrict__ Q, const __bf16* __restrict__ Kd,
                 const __bf16* __restrict__ Vt, const int* __restrict__ mask,
                 __bf16* __restrict__ Ctx) {
    __shared__ __bf16 plds[16 * 32];

    const int lane = threadIdx.x & 31;
    const int q0 = blockIdx.x * 16;
    const int h  = blockIdx.y;
    const int b  = blockIdx.z;

    const int ml   = lane & 15;
    const int kb8  = (lane & 16) ? 8  : 0;
    const int kb16 = (lane & 16) ? 16 : 0;
    const int hi8  = (lane & 16) ? 8  : 0;

    const float SCALE = 0.125f;                       // 1/sqrt(64)
    const float L2E   = 1.44269504088896340736f;

    // Q fragments: d 0..31 and d 32..63
    const __bf16* qbase = Q + ((size_t)(b * SS + q0 + ml)) * D_MODEL + h * HD;
    v16bf aq0 = load_a_frag(qbase + kb16);
    v16bf aq1 = load_a_frag(qbase + 32 + kb16);

    float mrow[8], lrow[8];
    v8f o[4];
#pragma unroll
    for (int r = 0; r < 8; ++r) { mrow[r] = -1e30f; lrow[r] = 0.f; }
#pragma unroll
    for (int f = 0; f < 4; ++f)
#pragma unroll
        for (int r = 0; r < 8; ++r) o[f][r] = 0.f;

    for (int t0 = 0; t0 < SS; t0 += 32) {
        // ---- S = Q K^T  (16 x 32, two 16x16 C tiles) ----
        v8f s0, s1;
#pragma unroll
        for (int r = 0; r < 8; ++r) { s0[r] = 0.f; s1[r] = 0.f; }

        const __bf16* kbp = Kd + ((size_t)(b * SS + t0 + ml)) * D_MODEL + h * HD;
        v16bf bk0a = *(const v16bf*)(kbp + kb16);               // cols t0..t0+15, d 0..31
        v16bf bk0b = *(const v16bf*)(kbp + 32 + kb16);          // d 32..63
        const __bf16* kbp1 = kbp + 16 * D_MODEL;                // cols t0+16..t0+31
        v16bf bk1a = *(const v16bf*)(kbp1 + kb16);
        v16bf bk1b = *(const v16bf*)(kbp1 + 32 + kb16);

        s0 = __builtin_amdgcn_wmma_f32_16x16x32_bf16(false, aq0, false, bk0a, (short)0, s0, false, false);
        s0 = __builtin_amdgcn_wmma_f32_16x16x32_bf16(false, aq1, false, bk0b, (short)0, s0, false, false);
        s1 = __builtin_amdgcn_wmma_f32_16x16x32_bf16(false, aq0, false, bk1a, (short)0, s1, false, false);
        s1 = __builtin_amdgcn_wmma_f32_16x16x32_bf16(false, aq1, false, bk1b, (short)0, s1, false, false);

        // ---- mask + online softmax (rows live in 16-lane halves) ----
        float al[8], p0v[8], p1v[8];
#pragma unroll
        for (int r = 0; r < 8; ++r) {
            int row = q0 + hi8 + r;
            const int* mp = mask + ((size_t)b * SS + row) * SS + t0;
            float e0 = s0[r] * SCALE;
            float e1 = s1[r] * SCALE;
            if (mp[ml] == 0)      e0 = NEGF;
            if (mp[16 + ml] == 0) e1 = NEGF;
            float pm = fmaxf(e0, e1);
#pragma unroll
            for (int off = 1; off < 16; off <<= 1)
                pm = fmaxf(pm, __shfl_xor(pm, off, 32));
            float mn = fmaxf(mrow[r], pm);
            float a  = exp2f((mrow[r] - mn) * L2E);
            float q0e = exp2f((e0 - mn) * L2E);
            float q1e = exp2f((e1 - mn) * L2E);
            float rs = q0e + q1e;
#pragma unroll
            for (int off = 1; off < 16; off <<= 1)
                rs += __shfl_xor(rs, off, 32);
            lrow[r] = lrow[r] * a + rs;
            mrow[r] = mn;
            al[r] = a; p0v[r] = q0e; p1v[r] = q1e;
        }

        // rescale running O by alpha
#pragma unroll
        for (int f = 0; f < 4; ++f)
#pragma unroll
            for (int r = 0; r < 8; ++r) o[f][r] *= al[r];

        // ---- C-layout P -> LDS -> A-layout fragment ----
#pragma unroll
        for (int r = 0; r < 8; ++r) {
            int row = hi8 + r;
            plds[row * 32 + ml]      = f2bf(p0v[r]);
            plds[row * 32 + 16 + ml] = f2bf(p1v[r]);
        }
        __syncthreads();
        v16bf ap = load_a_frag(&plds[ml * 32 + kb16]);
        __syncthreads();

        // ---- O += P @ V (4 feature tiles of 16) ----
#pragma unroll
        for (int f = 0; f < 4; ++f) {
            const __bf16* vp = Vt + ((size_t)b * D_MODEL + h * HD + f * 16 + ml) * SS + t0 + kb16;
            v16bf bv = *(const v16bf*)vp;
            o[f] = __builtin_amdgcn_wmma_f32_16x16x32_bf16(
                false, ap, false, bv, (short)0, o[f], false, false);
        }
    }

    // ---- epilogue: normalize and write context (bf16) ----
#pragma unroll
    for (int f = 0; f < 4; ++f)
#pragma unroll
        for (int r = 0; r < 8; ++r) {
            int row = q0 + hi8 + r;
            float val = o[f][r] / lrow[r];
            Ctx[((size_t)(b * SS + row)) * D_MODEL + h * HD + f * 16 + ml] = f2bf(val);
        }
}

// ---------------------------------------------------------------------------
// Host launcher
// ---------------------------------------------------------------------------
extern "C" void kernel_launch(void* const* d_in, const int* in_sizes, int n_in,
                              void* d_out, int out_size, void* d_ws, size_t ws_size,
                              hipStream_t stream) {
    (void)in_sizes; (void)n_in; (void)out_size; (void)ws_size;

    const float* x    = (const float*)d_in[0];
    const int*   mask = (const int*)  d_in[1];
    const float* Wq   = (const float*)d_in[2];
    const float* bq   = (const float*)d_in[3];
    const float* Wk   = (const float*)d_in[4];
    const float* bk   = (const float*)d_in[5];
    const float* Wv   = (const float*)d_in[6];
    const float* bv   = (const float*)d_in[7];
    const float* Wo   = (const float*)d_in[8];
    const float* bo   = (const float*)d_in[9];
    float* out = (float*)d_out;

    // Workspace layout (bf16 buffers), 72 MiB total:
    const size_t NX = (size_t)MROWS * D_MODEL;          // 8M elems
    const size_t NW = (size_t)D_MODEL * D_MODEL;        // 1M elems
    char* ws = (char*)d_ws;
    __bf16* X16 = (__bf16*)(ws);                        // 16 MiB (reused as Ctx)
    __bf16* WqT = (__bf16*)(ws + NX * 2);               //  2 MiB
    __bf16* WkT = WqT + NW;
    __bf16* WvT = WkT + NW;
    __bf16* WoT = WvT + NW;
    __bf16* Qb  = WoT + NW;                             // 16 MiB
    __bf16* Kb  = Qb + NX;                              // 16 MiB
    __bf16* Vt  = Kb + NX;                              // 16 MiB [B][D][S]
    __bf16* Ctx = X16;                                  // alias: x no longer needed

    // 1) convert activations + weights to bf16 (weights transposed to [N,K])
    cvt_f32_bf16<<<(int)(NX / 256), 256, 0, stream>>>(x, X16, (int)NX);
    transpose_cvt<<<(int)(NW / 256), 256, 0, stream>>>(Wq, WqT);
    transpose_cvt<<<(int)(NW / 256), 256, 0, stream>>>(Wk, WkT);
    transpose_cvt<<<(int)(NW / 256), 256, 0, stream>>>(Wv, WvT);
    transpose_cvt<<<(int)(NW / 256), 256, 0, stream>>>(Wo, WoT);

    // 2) projections (WMMA GEMMs). V is written transposed per head.
    dim3 ggrid(D_MODEL / 64, MROWS / 256);  // (16, 32)
    gemm_bf16<__bf16, false><<<ggrid, 256, 0, stream>>>(X16, WqT, bq, Qb, MROWS, D_MODEL, D_MODEL);
    gemm_bf16<__bf16, false><<<ggrid, 256, 0, stream>>>(X16, WkT, bk, Kb, MROWS, D_MODEL, D_MODEL);
    gemm_bf16<__bf16, true ><<<ggrid, 256, 0, stream>>>(X16, WvT, bv, Vt, MROWS, D_MODEL, D_MODEL);

    // 3) flash attention -> context (bf16)
    dim3 agrid(SS / 16, NH, BB);            // (128, 16, 4)
    attn_kernel<<<agrid, 32, 0, stream>>>(Qb, Kb, Vt, mask, Ctx);

    // 4) output projection (fp32 out)
    gemm_bf16<float, false><<<ggrid, 256, 0, stream>>>(Ctx, WoT, bo, out, MROWS, D_MODEL, D_MODEL);
}
